// RNN_LowRank_188978561253
// MI455X (gfx1250) — matrix operand order repaired and checked by reading, not compiled, not run on hardware
//
#include <hip/hip_runtime.h>

typedef __attribute__((ext_vector_type(16))) _Float16 v16h;
typedef __attribute__((ext_vector_type(8)))  float    v8f;

#define HIDDEN      4096
#define RANK        8
#define NIN         4
#define NOUT        3
#define TLEN        3072
#define ALPHA       0.2f
#define NOISE_SCALE 0.07071067811865475f  // sqrt(2)*0.05

struct alignas(16) H8 { _Float16 v[8]; };   // one ds_load_b128 of f16 h-state
struct alignas(8)  H4 { _Float16 v[4]; };   // per-thread h16 store (ds_store_b64)

// ---------------------------------------------------------------------------
// Zero-fill d_out (rows with t >= end must be exactly 0; harness poisons it).
// ---------------------------------------------------------------------------
__global__ void rnn_zero_kernel(float* __restrict__ out, int n) {
    int idx    = blockIdx.x * blockDim.x + threadIdx.x;
    int stride = gridDim.x * blockDim.x;
    for (int i = idx; i < n; i += stride) out[i] = 0.0f;
}

// Shared WMMA phase: 4x v_wmma_f32_16x16x32_f16 over this wave's 128-wide
// slice of hsh against the resident B fragments ([N;W_out] packed columns).
// A 16x32 f16 layout: every row = same h chunk (broadcast matvec);
// lanes 0-15 need K{0..7,16..23}, lanes 16-31 need K{8..15,24..31}.
__device__ __forceinline__
v8f wmma_phase(const _Float16* hsh, const v16h* Bf, int wave, int lane) {
    v8f cacc;
    #pragma unroll
    for (int e = 0; e < 8; ++e) cacc[e] = 0.0f;
    #pragma unroll
    for (int c = 0; c < 4; ++c) {
        const int base = wave * 128 + c * 32 + ((lane & 16) ? 8 : 0);
        const H8* p = (const H8*)&hsh[base];
        union { v16h v; H8 s[2]; } au;
        au.s[0] = p[0];       // K = kbase .. kbase+7      (ds_load_b128)
        au.s[1] = p[2];       // K = kbase+16 .. kbase+23  (ds_load_b128)
        cacc = __builtin_amdgcn_wmma_f32_16x16x32_f16(
                   false, au.v, false, Bf[c], (short)0, cacc, false, false);
    }
    return cacc;
}

// ---------------------------------------------------------------------------
// Persistent single-workgroup sequential scan. 1024 threads = 32 wave32 waves.
// B packs [N (8 rows); W_out (3 rows)] so the same 4 WMMAs per step produce
// BOTH r(t) = N@h_state AND the previous step's readout o(t-1) = W_out@h_state.
// Per step: noise float4 + uniform u float4 loads issued first (latency hidden
// behind the WMMA phase); branchless partial stores; 11-column reducer; then
// the elementwise softplus update. 3 barriers/step (minimum for the dataflow).
// ---------------------------------------------------------------------------
__global__ __launch_bounds__(1024, 1)
void rnn_lowrank_scan_kernel(const float* __restrict__ input,
                             const int*   __restrict__ ends,
                             const int*   __restrict__ bptr,
                             const float* __restrict__ noise,
                             const float* __restrict__ M,
                             const float* __restrict__ N,
                             const float* __restrict__ Win,
                             const float* __restrict__ Wout,
                             float* __restrict__ out) {
    __shared__ alignas(16) _Float16 hsh[HIDDEN];   // h state as f16 (WMMA A src)
    __shared__ float allpart[32 * 32];             // branchless per-wave partials
    __shared__ float rvec[RANK];                   // reduced r

    const int tid  = threadIdx.x;
    const int lane = tid & 31;
    const int wave = tid >> 5;
    const int i0   = tid * 4;                      // first owned hidden unit

    const int b = bptr[0];
    int end = ends[b];                             // b==0: safe for i32/i64 backing
    if (end > TLEN) end = TLEN;

    // ---- loop-invariant register staging ----------------------------------
    // Combined matrix rows: 0..7 = N, 8..10 = W_out, 11..15 = zero pad.
    // 16-bit B 32x16 layout: lane = column (mod 16); lanes>=16 hold K{8..15,24..31};
    // vector element e maps to K-row khalf+e (e<8) or khalf+e+8 (e>=8).
    v16h Bf[4];
    {
        const int col   = lane & 15;
        const int khalf = (lane & 16) ? 8 : 0;
        #pragma unroll
        for (int c = 0; c < 4; ++c) {
            const int base = wave * 128 + c * 32 + khalf;
            #pragma unroll
            for (int e = 0; e < 16; ++e) {
                float x = 0.0f;
                if (col < RANK)             x = N[col * HIDDEN + base + (e < 8 ? e : e + 8)];
                else if (col < RANK + NOUT) x = Wout[(size_t)(col - RANK) * HIDDEN + base + (e < 8 ? e : e + 8)];
                Bf[c][e] = (_Float16)x;
            }
        }
    }
    float Mr[4][RANK], Wi[4][NIN], h[4];
    #pragma unroll
    for (int j = 0; j < 4; ++j) {
        #pragma unroll
        for (int k = 0; k < RANK; ++k) Mr[j][k] = M[(size_t)(i0 + j) * RANK + k];
        #pragma unroll
        for (int k = 0; k < NIN; ++k)  Wi[j][k] = Win[(size_t)(i0 + j) * NIN + k];
        h[j] = 0.0f;
        hsh[i0 + j] = (_Float16)0.0f;
    }

    // Running row pointers (strength-reduced; one add per step each).
    const float* nrow = noise + i0;                          // noise[t, i0..]
    float*       hrow = out + (size_t)NOUT * TLEN + i0;      // hidden[t, i0..]
    const float* urow = input + (size_t)b * TLEN * NIN;      // u[t, :] (uniform)
    __syncthreads();

    for (int t = 0; t < end; ++t) {
        // Early loads for this step: per-thread noise float4 plus a
        // wave-uniform u_t float4 (same cacheline for all lanes -> broadcast;
        // no LDS staging, no divergent path). Latency hides behind phase A.
        const float4 nz = *(const float4*)nrow;
        const float4 uv = *(const float4*)urow;

        // ---- phase A: [r(t); o(t-1)] partials via WMMA --------------------
        const v8f cacc = wmma_phase(hsh, Bf, wave, lane);
        // C row 0 = element 0, lanes 0-15 = columns of B = combined rows.
        // Branchless: every lane stores; reducers read cols 0..10 only.
        allpart[wave * 32 + lane] = cacc[0];
        // Speculative prefetch of next noise row (global_prefetch_b8;
        // one-past-end at the final step is dropped harmlessly).
        __builtin_prefetch(nrow + HIDDEN, 0, 1);
        __syncthreads();  // B1

        // ---- merged tiny reduction (threads 0-10, one exec mask) ----------
        if (tid < RANK + NOUT) {
            float s = 0.0f;
            #pragma unroll
            for (int w = 0; w < 32; ++w) s += allpart[w * 32 + tid];
            if (tid < RANK) {
                rvec[tid] = s;
            } else if (t > 0) {
                s = fminf(fmaxf(s, -1000.0f), 1000.0f);
                out[(size_t)(t - 1) * NOUT + (tid - RANK)] = s;  // o(t-1)
            }
        }
        __syncthreads();  // B2

        // ---- phase D: elementwise state update ----------------------------
        float rv[RANK];
        #pragma unroll
        for (int k = 0; k < RANK; ++k) rv[k] = rvec[k];
        const float uu[4]  = {uv.x, uv.y, uv.z, uv.w};
        const float nzv[4] = {nz.x, nz.y, nz.z, nz.w};

        float hv[4];
        #pragma unroll
        for (int j = 0; j < 4; ++j) {
            float mr = 0.0f;
            #pragma unroll
            for (int k = 0; k < RANK; ++k) mr += Mr[j][k] * rv[k];
            float wu = 0.0f;
            #pragma unroll
            for (int k = 0; k < NIN; ++k) wu += Wi[j][k] * uu[k];
            const float I   = mr * (1.0f / (float)HIDDEN) + wu + NOISE_SCALE * nzv[j];
            const float pre = (1.0f - ALPHA) * h[j] + ALPHA * I;
            // numerically stable softplus: max(x,0) + log1p(exp(-|x|))
            const float hn = fmaxf(pre, 0.0f) + log1pf(__expf(-fabsf(pre)));
            h[j]  = hn;
            hv[j] = hn;
        }
        // hidden_all[t] = h_new (active region only; tail pre-zeroed)
        *(float4*)hrow = make_float4(hv[0], hv[1], hv[2], hv[3]);
        // refresh f16 state for next step's WMMA A fragment (ds_store_b64)
        H4 hh;
        #pragma unroll
        for (int j = 0; j < 4; ++j) hh.v[j] = (_Float16)hv[j];
        *(H4*)&hsh[i0] = hh;

        nrow += HIDDEN;
        hrow += HIDDEN;
        urow += NIN;
        __syncthreads();  // B3: hsh(t) visible to phase A of t+1
    }

    // ---- epilogue: emit out[end-1] = clip(W_out @ h_new(end-1)) -----------
    if (end > 0) {
        const v8f cacc = wmma_phase(hsh, Bf, wave, lane);
        allpart[wave * 32 + lane] = cacc[0];
        __syncthreads();
        if (tid >= RANK && tid < RANK + NOUT) {
            float s = 0.0f;
            #pragma unroll
            for (int w = 0; w < 32; ++w) s += allpart[w * 32 + tid];
            s = fminf(fmaxf(s, -1000.0f), 1000.0f);
            out[(size_t)(end - 1) * NOUT + (tid - RANK)] = s;
        }
    }
}

// ---------------------------------------------------------------------------
extern "C" void kernel_launch(void* const* d_in, const int* in_sizes, int n_in,
                              void* d_out, int out_size, void* d_ws, size_t ws_size,
                              hipStream_t stream) {
    const float* input = (const float*)d_in[0];   // (BATCH, T, N_IN) f32
    const int*   ends  = (const int*)  d_in[1];   // (1, BATCH) int
    const int*   bptr  = (const int*)  d_in[2];   // scalar b
    const float* noise = (const float*)d_in[3];   // (T, H) f32
    const float* M     = (const float*)d_in[4];   // (H, RANK) f32
    const float* N     = (const float*)d_in[5];   // (RANK, H) f32
    const float* Win   = (const float*)d_in[6];   // (H, N_IN) f32
    const float* Wout  = (const float*)d_in[7];   // (N_OUT, H) f32
    float* out = (float*)d_out;                   // [T*3 | T*H] f32

    rnn_zero_kernel<<<2048, 256, 0, stream>>>(out, out_size);
    rnn_lowrank_scan_kernel<<<1, 1024, 0, stream>>>(input, ends, bptr, noise,
                                                    M, N, Win, Wout, out);
}
